// TransMIL_72859825209793
// MI455X (gfx1250) — matrix operand order repaired and checked by reading, not compiled, not run on hardware
//
#include <hip/hip_runtime.h>

// ---------------- problem constants ----------------
#define DIMC   512
#define HEADS  8
#define DH     64
#define LANDC  256
#define NSEQ   16000
#define KIN    1024
#define HWDIM  127
#define NFEAT  16129            // 127*127
#define NTOK   16130            // 1 + NFEAT
#define NPAD   16384            // padded nystrom length
#define PADR   254              // zero rows prepended
#define LSEG   64               // NPAD / LANDC

typedef __attribute__((ext_vector_type(16))) __bf16 v16bf;
typedef __attribute__((ext_vector_type(8)))  __bf16 bf16x8;
typedef __attribute__((ext_vector_type(8)))  float  v8f;

static __device__ __forceinline__ v16bf cat8(bf16x8 lo, bf16x8 hi) {
  v16bf r;
#pragma unroll
  for (int i = 0; i < 8; ++i) { r[i] = lo[i]; r[i + 8] = hi[i]; }
  return r;
}

// Async global->LDS 16B copy (CDNA5 GLOBAL_LOAD_ASYNC_TO_LDS_B128, ASYNCcnt).
static __device__ __forceinline__ void async_b128(const void* gptr, void* lptr) {
  unsigned lds = (unsigned)(size_t)lptr;                 // LDS address: low 32 bits of the pointer
  unsigned long long ga = (unsigned long long)(size_t)gptr;
  asm volatile("global_load_async_to_lds_b128 %0, %1, off" :: "v"(lds), "v"(ga) : "memory");
}
static __device__ __forceinline__ void wait_async0() {
  asm volatile("s_wait_asynccnt 0" ::: "memory");
}

// ================= generic batched bf16 WMMA GEMM =================
// C[b] = act( A[b](MxK,row-major) * B[b] + bias )
// B layout: bKmajor==0 -> memory is (N x K) row-major ("weights", i.e. B^T)
//           bKmajor==1 -> memory is (K x N) row-major
// Requires: M%128==0, N%64==0, K%32==0 (true for every call site here).
// Pipeline: double-buffered LDS tiles staged with async-to-LDS, overlapped
// with the 2x2 v_wmma_f32_16x16x32_bf16 compute of the previous tile.
__global__ __launch_bounds__(256)
void gemm_bf16(const __bf16* __restrict__ A, const __bf16* __restrict__ B,
               float* __restrict__ Cf, __bf16* __restrict__ Cb,
               const float* __restrict__ bias,
               int M, int N, int K,
               long long sA, long long sB, long long sC,
               int bKmajor, int relu)
{
  __shared__ __bf16 As[2][128 * 40];
  __shared__ __bf16 Bs[2][64 * 40];
  const int tid  = threadIdx.x;
  const int lane = tid & 31;
  const int wave = tid >> 5;
  const int wr = wave & 3;          // 4 wave-rows of 32
  const int wc = wave >> 2;         // 2 wave-cols of 32
  const int n0 = blockIdx.x * 64;
  const long long m0 = (long long)blockIdx.y * 128;
  const __bf16* Ab = A + (long long)blockIdx.z * sA + m0 * K;
  const __bf16* Bb = B + (long long)blockIdx.z * sB;

  // staging coordinates (8 halves = 16B per thread per pass)
  const int ar = (tid * 8) >> 5;    // 0..63
  const int ac = (tid * 8) & 31;
  const int kk = (tid * 8) >> 6;    // kmajor: 0..31
  const int nn = (tid * 8) & 63;

  v8f acc[2][2] = {};

  auto stage = [&](int k0, int buf) {
    // A tile 128x32: two async b128 per thread
#pragma unroll
    for (int p = 0; p < 2; ++p) {
      int r = ar + p * 64;
      async_b128(Ab + (long long)r * K + (k0 + ac), &As[buf][r * 40 + ac]);
    }
    // B tile -> LDS as [n][k] (column-major B)
    if (!bKmajor) {
      async_b128(Bb + (long long)(n0 + ar) * K + (k0 + ac), &Bs[buf][ar * 40 + ac]);
    } else {
      bf16x8 v = *(const bf16x8*)(Bb + (long long)(k0 + kk) * N + (n0 + nn));
#pragma unroll
      for (int i = 0; i < 8; ++i) Bs[buf][(nn + i) * 40 + kk] = v[i];
    }
  };

  stage(0, 0);
  wait_async0();
  __syncthreads();

  int buf = 0;
  for (int k0 = 0; k0 < K; k0 += 32) {
    if (k0 + 32 < K) stage(k0 + 32, buf ^ 1);   // overlap next tile with compute

    const int g = lane >> 4, lr = lane & 15;
    v16bf afr[2], bfr[2];
#pragma unroll
    for (int mi = 0; mi < 2; ++mi) {
      int row = wr * 32 + mi * 16 + lr;          // A: lane holds K {8g..8g+7} U {16+8g..}
      afr[mi] = cat8(*(const bf16x8*)(&As[buf][row * 40 + 8 * g]),
                     *(const bf16x8*)(&As[buf][row * 40 + 16 + 8 * g]));
    }
#pragma unroll
    for (int ni = 0; ni < 2; ++ni) {
      int col = wc * 32 + ni * 16 + lr;          // B: lane holds K {8g..8g+7} U {16+8g..}
      bfr[ni] = cat8(*(const bf16x8*)(&Bs[buf][col * 40 + 8 * g]),
                     *(const bf16x8*)(&Bs[buf][col * 40 + 16 + 8 * g]));
    }
#pragma unroll
    for (int mi = 0; mi < 2; ++mi)
#pragma unroll
      for (int ni = 0; ni < 2; ++ni)
        acc[mi][ni] = __builtin_amdgcn_wmma_f32_16x16x32_bf16(
            false, afr[mi], false, bfr[ni], (short)0, acc[mi][ni], false, false);

    wait_async0();          // this wave's async stores to buf^1 landed
    __syncthreads();        // all waves done reading buf / staging buf^1
    buf ^= 1;
  }

  // ---- epilogue: C layout lane=N%16, VGPR r -> M = r + 8*(lane>=16) ----
  const int g = lane >> 4, lr = lane & 15;
#pragma unroll
  for (int mi = 0; mi < 2; ++mi)
#pragma unroll
    for (int ni = 0; ni < 2; ++ni) {
      int col = n0 + wc * 32 + ni * 16 + lr;
      long long cbase = (long long)blockIdx.z * sC + col;
#pragma unroll
      for (int r = 0; r < 8; ++r) {
        long long rowg = m0 + wr * 32 + mi * 16 + r + 8 * g;
        float v = acc[mi][ni][r];
        if (bias) v += bias[col];
        if (relu) v = fmaxf(v, 0.f);
        long long off = cbase + rowg * N;
        if (Cf) Cf[off] = v;
        if (Cb) Cb[off] = (__bf16)v;
      }
    }
}

// ================= elementwise / reduction kernels =================
__global__ __launch_bounds__(256)
void cvt_bf16(const float* __restrict__ x, __bf16* __restrict__ y, long long n) {
  long long i = (long long)blockIdx.x * 256 + threadIdx.x;
  if (i < n) y[i] = (__bf16)x[i];
}

__global__ __launch_bounds__(256)
void assemble_tok(const float* __restrict__ h, const float* __restrict__ cls,
                  float* __restrict__ tok) {
  long long t = (long long)blockIdx.x * 256 + threadIdx.x;
  if (t >= (long long)NTOK * DIMC) return;
  int c = (int)(t & (DIMC - 1));
  long long r = t >> 9;
  float v;
  if (r == 0) v = cls[c];
  else { long long i = r - 1; if (i >= NSEQ) i -= NSEQ; v = h[i * DIMC + c]; }
  tok[t] = v;
}

// layernorm rows of 512; writes bf16 into padded (NPAD x 512) buffer, pad rows = 0
__global__ __launch_bounds__(256)
void layernorm_pad(const float* __restrict__ X, const float* __restrict__ g,
                   const float* __restrict__ b, __bf16* __restrict__ O) {
  int r = blockIdx.x, tid = threadIdx.x;
  __bf16* o = O + (long long)r * DIMC;
  if (r < PADR) { o[tid] = (__bf16)0.f; o[tid + 256] = (__bf16)0.f; return; }
  const float* x = X + (long long)(r - PADR) * DIMC;
  float a0 = x[tid], a1 = x[tid + 256];
  __shared__ float red[256];
  red[tid] = a0 + a1; __syncthreads();
  for (int s = 128; s > 0; s >>= 1) { if (tid < s) red[tid] += red[tid + s]; __syncthreads(); }
  float mu = red[0] * (1.f / DIMC); __syncthreads();
  float d0 = a0 - mu, d1 = a1 - mu;
  red[tid] = d0 * d0 + d1 * d1; __syncthreads();
  for (int s = 128; s > 0; s >>= 1) { if (tid < s) red[tid] += red[tid + s]; __syncthreads(); }
  float rstd = rsqrtf(red[0] * (1.f / DIMC) + 1e-5f);
  o[tid]       = (__bf16)(d0 * rstd * g[tid] + b[tid]);
  o[tid + 256] = (__bf16)(d1 * rstd * g[tid + 256] + b[tid + 256]);
}

// qkv row-major (NPAD x 1536) -> per-head q(scaled)/k/v (HEADS x NPAD x 64) bf16
__global__ __launch_bounds__(256)
void split_heads(const __bf16* __restrict__ qkv, __bf16* __restrict__ qh,
                 __bf16* __restrict__ kh, __bf16* __restrict__ vh) {
  long long t = (long long)blockIdx.x * 256 + threadIdx.x;
  if (t >= (long long)NPAD * DIMC) return;
  int col = (int)(t & (DIMC - 1));
  long long row = t >> 9;
  int h = col >> 6, d = col & 63;
  long long o = (long long)h * NPAD * DH + row * DH + d;
  const __bf16* r = qkv + row * (3 * DIMC);
  qh[o] = (__bf16)((float)r[col] * 0.125f);  // DHEAD^-0.5
  kh[o] = r[DIMC + col];
  vh[o] = r[2 * DIMC + col];
}

// landmarks: mean of 64 consecutive rows -> (HEADS x 256 x 64)
__global__ __launch_bounds__(256)
void landmarks(const __bf16* __restrict__ X, __bf16* __restrict__ Lm) {
  int t = blockIdx.x * 256 + threadIdx.x;
  if (t >= HEADS * LANDC * DH) return;
  int d = t & 63, j = (t >> 6) & 255, h = t >> 14;
  const __bf16* x = X + (long long)h * NPAD * DH + (long long)j * LSEG * DH + d;
  float s = 0.f;
  for (int i = 0; i < LSEG; ++i) s += (float)x[i * DH];
  Lm[t] = (__bf16)(s * (1.f / LSEG));
}

__global__ __launch_bounds__(256)
void softmax_rows(const float* __restrict__ X, __bf16* __restrict__ Ob,
                  float* __restrict__ Of, int len) {
  long long row = blockIdx.x;
  const float* xr = X + row * (long long)len;
  int tid = threadIdx.x;
  __shared__ float red[256];
  float mx = -3.0e38f;
  for (int i = tid; i < len; i += 256) mx = fmaxf(mx, xr[i]);
  red[tid] = mx; __syncthreads();
  for (int s = 128; s > 0; s >>= 1) { if (tid < s) red[tid] = fmaxf(red[tid], red[tid + s]); __syncthreads(); }
  mx = red[0]; __syncthreads();
  float sum = 0.f;
  for (int i = tid; i < len; i += 256) sum += __expf(xr[i] - mx);
  red[tid] = sum; __syncthreads();
  for (int s = 128; s > 0; s >>= 1) { if (tid < s) red[tid] += red[tid + s]; __syncthreads(); }
  float inv = 1.f / red[0];
  for (int i = tid; i < len; i += 256) {
    float v = __expf(xr[i] - mx) * inv;
    if (Ob) Ob[row * (long long)len + i] = (__bf16)v;
    if (Of) Of[row * (long long)len + i] = v;
  }
}

// global max over all heads of column-sums of P (softmax output, nonneg); rowsum max == 1
__global__ __launch_bounds__(256)
void pinv_gmax(const float* __restrict__ P, float* __restrict__ gmax) {
  int tid = threadIdx.x;
  __shared__ float red[256];
  float mx = 0.f;
  for (int cidx = tid; cidx < HEADS * LANDC; cidx += 256) {
    int h = cidx >> 8, n = cidx & 255;
    const float* p = P + (long long)h * LANDC * LANDC + n;
    float s = 0.f;
    for (int m = 0; m < LANDC; ++m) s += p[(long long)m * LANDC];
    mx = fmaxf(mx, s);
  }
  red[tid] = mx; __syncthreads();
  for (int s = 128; s > 0; s >>= 1) { if (tid < s) red[tid] = fmaxf(red[tid], red[tid + s]); __syncthreads(); }
  if (tid == 0) gmax[0] = red[0];
}

__global__ __launch_bounds__(256)
void pinv_init(const float* __restrict__ P, const float* __restrict__ gmax,
               float* __restrict__ Z) {
  long long t = (long long)blockIdx.x * 256 + threadIdx.x;
  if (t >= (long long)HEADS * LANDC * LANDC) return;
  int m = (int)(t & 255), n = (int)((t >> 8) & 255), h = (int)(t >> 16);
  Z[t] = P[(long long)h * LANDC * LANDC + (long long)m * LANDC + n] / gmax[0];
}

// C = s*I - A (per-head 256x256)
__global__ __launch_bounds__(256)
void smi_k(const float* __restrict__ A, float* __restrict__ C, float s) {
  long long t = (long long)blockIdx.x * 256 + threadIdx.x;
  if (t >= (long long)HEADS * LANDC * LANDC) return;
  int j = (int)(t & 255), i = (int)((t >> 8) & 255);
  C[t] = ((i == j) ? s : 0.f) - A[t];
}

// naive fp32 batched GEMM for the tiny pinv matrices
__global__ __launch_bounds__(256)
void mm_f32(const float* __restrict__ A, const float* __restrict__ B,
            float* __restrict__ C, int M, int N, int K, float alpha,
            long long sA, long long sB, long long sC) {
  int n = blockIdx.x * 16 + threadIdx.x;
  int m = blockIdx.y * 16 + threadIdx.y;
  const float* a = A + (long long)blockIdx.z * sA + (long long)m * K;
  const float* b = B + (long long)blockIdx.z * sB + n;
  float s = 0.f;
  for (int k = 0; k < K; ++k) s += a[k] * b[(long long)k * N];
  C[(long long)blockIdx.z * sC + (long long)m * N + n] = alpha * s;
}

// attn(h,row,d) + depthwise 33-tap conv over seq of v -> row-major bf16 (NPAD x 512)
__global__ __launch_bounds__(256)
void res_merge(const float* __restrict__ attn, const __bf16* __restrict__ vh,
               const float* __restrict__ rw, __bf16* __restrict__ outb) {
  long long t = (long long)blockIdx.x * 256 + threadIdx.x;
  if (t >= (long long)NPAD * DIMC) return;
  int col = (int)(t & (DIMC - 1));
  long long row = t >> 9;
  int h = col >> 6, d = col & 63;
  const __bf16* v = vh + (long long)h * NPAD * DH;
  float s = attn[(long long)h * NPAD * DH + row * DH + d];
#pragma unroll
  for (int k = 0; k < 33; ++k) {
    long long r2 = row + k - 16;
    if (r2 >= 0 && r2 < NPAD) s += (float)v[r2 * DH + d] * rw[h * 33 + k];
  }
  outb[t] = (__bf16)s;
}

__global__ __launch_bounds__(256)
void resid_tail(float* __restrict__ tok, const float* __restrict__ proj) {
  long long t = (long long)blockIdx.x * 256 + threadIdx.x;
  if (t >= (long long)NTOK * DIMC) return;
  tok[t] += proj[(long long)PADR * DIMC + t];  // out[:, -n:]
}

// PPEG: identity + depthwise 7x7 + 5x5 + 3x3 over (512,127,127); cls row passthrough
__global__ __launch_bounds__(256)
void ppeg_k(const float* __restrict__ tin, float* __restrict__ tout,
            const float* __restrict__ w7, const float* __restrict__ b7,
            const float* __restrict__ w5, const float* __restrict__ b5,
            const float* __restrict__ w3, const float* __restrict__ b3) {
  long long t = (long long)blockIdx.x * 256 + threadIdx.x;
  if (t >= (long long)NTOK * DIMC) return;
  int c = (int)(t & (DIMC - 1));
  long long r = t >> 9;
  if (r == 0) { tout[t] = tin[t]; return; }
  long long pos = r - 1;
  int y = (int)(pos / HWDIM), x = (int)(pos % HWDIM);
  float acc = tin[t] + b7[c] + b5[c] + b3[c];
  const float* wc = w7 + c * 49;
  for (int dy = 0; dy < 7; ++dy) {
    int yy = y + dy - 3; if ((unsigned)yy >= HWDIM) continue;
    for (int dx = 0; dx < 7; ++dx) {
      int x2 = x + dx - 3; if ((unsigned)x2 >= HWDIM) continue;
      acc += wc[dy * 7 + dx] * tin[(long long)(1 + yy * HWDIM + x2) * DIMC + c];
    }
  }
  wc = w5 + c * 25;
  for (int dy = 0; dy < 5; ++dy) {
    int yy = y + dy - 2; if ((unsigned)yy >= HWDIM) continue;
    for (int dx = 0; dx < 5; ++dx) {
      int x2 = x + dx - 2; if ((unsigned)x2 >= HWDIM) continue;
      acc += wc[dy * 5 + dx] * tin[(long long)(1 + yy * HWDIM + x2) * DIMC + c];
    }
  }
  wc = w3 + c * 9;
  for (int dy = 0; dy < 3; ++dy) {
    int yy = y + dy - 1; if ((unsigned)yy >= HWDIM) continue;
    for (int dx = 0; dx < 3; ++dx) {
      int x2 = x + dx - 1; if ((unsigned)x2 >= HWDIM) continue;
      acc += wc[dy * 3 + dx] * tin[(long long)(1 + yy * HWDIM + x2) * DIMC + c];
    }
  }
  tout[t] = acc;
}

// final: layernorm(cls row) then fc2 (4 x 512)
__global__ __launch_bounds__(256)
void final_head(const float* __restrict__ tok, const float* __restrict__ g,
                const float* __restrict__ b, const float* __restrict__ w,
                const float* __restrict__ bias, float* __restrict__ out) {
  int tid = threadIdx.x;
  __shared__ float red[256];
  __shared__ float y[DIMC];
  float a0 = tok[tid], a1 = tok[tid + 256];
  red[tid] = a0 + a1; __syncthreads();
  for (int s = 128; s > 0; s >>= 1) { if (tid < s) red[tid] += red[tid + s]; __syncthreads(); }
  float mu = red[0] * (1.f / DIMC); __syncthreads();
  float d0 = a0 - mu, d1 = a1 - mu;
  red[tid] = d0 * d0 + d1 * d1; __syncthreads();
  for (int s = 128; s > 0; s >>= 1) { if (tid < s) red[tid] += red[tid + s]; __syncthreads(); }
  float rstd = rsqrtf(red[0] * (1.f / DIMC) + 1e-5f);
  y[tid]       = d0 * rstd * g[tid] + b[tid];
  y[tid + 256] = d1 * rstd * g[tid + 256] + b[tid + 256];
  __syncthreads();
  int j = tid >> 6, l = tid & 63;
  float s = 0.f;
  for (int i = l; i < DIMC; i += 64) s += y[i] * w[j * DIMC + i];
  red[tid] = s; __syncthreads();
  for (int st = 32; st > 0; st >>= 1) { if (l < st) red[tid] += red[tid + st]; __syncthreads(); }
  if (l == 0) out[j] = red[tid] + bias[j];
}

// ================= host orchestration =================
static void gemmL(hipStream_t st, const __bf16* A, const __bf16* B,
                  float* Cf, __bf16* Cb, const float* bias,
                  int M, int N, int K, long long sA, long long sB, long long sC,
                  int batch, int bKmajor, int relu) {
  dim3 g(N / 64, M / 128, batch), blk(256);
  gemm_bf16<<<g, blk, 0, st>>>(A, B, Cf, Cb, bias, M, N, K, sA, sB, sC, bKmajor, relu);
}

struct LayerBufs {
  __bf16 *xln, *qh, *kh, *vh, *qlb, *klb, *a1b, *a3b, *pvb, *attnb, *qkvo;
  float *Sbig, *s2, *z0, *z1, *xz, *tb, *ub, *gmax, *av, *pv, *attn, *proj;
};

static void run_layer(hipStream_t st, float* tok,
                      const float* ng, const float* nb,
                      const __bf16* wqkv, const __bf16* wout,
                      const float* outb, const float* resw, const LayerBufs& B) {
  layernorm_pad<<<NPAD, 256, 0, st>>>(tok, ng, nb, B.xln);
  gemmL(st, B.xln, wqkv, nullptr, B.qkvo, nullptr, NPAD, 3 * DIMC, DIMC, 0, 0, 0, 1, 0, 0);
  split_heads<<<(NPAD * DIMC) / 256, 256, 0, st>>>(B.qkvo, B.qh, B.kh, B.vh);
  landmarks<<<(HEADS * LANDC * DH) / 256, 256, 0, st>>>(B.qh, B.qlb);
  landmarks<<<(HEADS * LANDC * DH) / 256, 256, 0, st>>>(B.kh, B.klb);
  // a1 = softmax(q @ kl^T)    [16384 x 256 per head]
  gemmL(st, B.qh, B.klb, B.Sbig, nullptr, nullptr, NPAD, LANDC, DH,
        (long long)NPAD * DH, (long long)LANDC * DH, (long long)NPAD * LANDC, HEADS, 0, 0);
  softmax_rows<<<HEADS * NPAD, 256, 0, st>>>(B.Sbig, B.a1b, nullptr, LANDC);
  // a3 = softmax(ql @ k^T)    [256 x 16384 per head]
  gemmL(st, B.qlb, B.kh, B.Sbig, nullptr, nullptr, LANDC, NPAD, DH,
        (long long)LANDC * DH, (long long)NPAD * DH, (long long)LANDC * NPAD, HEADS, 0, 0);
  softmax_rows<<<HEADS * LANDC, 256, 0, st>>>(B.Sbig, B.a3b, nullptr, NPAD);
  // a2 = softmax(ql @ kl^T)   [256 x 256 per head], fp32 in place
  gemmL(st, B.qlb, B.klb, B.s2, nullptr, nullptr, LANDC, LANDC, DH,
        (long long)LANDC * DH, (long long)LANDC * DH, (long long)LANDC * LANDC, HEADS, 0, 0);
  softmax_rows<<<HEADS * LANDC, 256, 0, st>>>(B.s2, nullptr, B.s2, LANDC);
  // Newton-Schulz pseudo-inverse of a2
  const long long S = (long long)LANDC * LANDC;
  pinv_gmax<<<1, 256, 0, st>>>(B.s2, B.gmax);
  pinv_init<<<(HEADS * (int)S) / 256, 256, 0, st>>>(B.s2, B.gmax, B.z0);
  dim3 g16(16, 16, HEADS), b16(16, 16);
  float *z = B.z0, *zn = B.z1;
  for (int it = 0; it < 6; ++it) {
    mm_f32<<<g16, b16, 0, st>>>(B.s2, z, B.xz, LANDC, LANDC, LANDC, 1.f, S, S, S);
    smi_k<<<(HEADS * (int)S) / 256, 256, 0, st>>>(B.xz, B.tb, 7.f);
    mm_f32<<<g16, b16, 0, st>>>(B.xz, B.tb, B.ub, LANDC, LANDC, LANDC, 1.f, S, S, S);
    smi_k<<<(HEADS * (int)S) / 256, 256, 0, st>>>(B.ub, B.tb, 15.f);
    mm_f32<<<g16, b16, 0, st>>>(B.xz, B.tb, B.ub, LANDC, LANDC, LANDC, 1.f, S, S, S);
    smi_k<<<(HEADS * (int)S) / 256, 256, 0, st>>>(B.ub, B.tb, 13.f);
    mm_f32<<<g16, b16, 0, st>>>(z, B.tb, zn, LANDC, LANDC, LANDC, 0.25f, S, S, S);
    float* tmp = z; z = zn; zn = tmp;
  }
  // av = a3 @ v, pv = pinv @ av, out = a1 @ pv
  gemmL(st, B.a3b, B.vh, B.av, nullptr, nullptr, LANDC, DH, NPAD,
        (long long)LANDC * NPAD, (long long)NPAD * DH, (long long)LANDC * DH, HEADS, 1, 0);
  mm_f32<<<dim3(DH / 16, LANDC / 16, HEADS), b16, 0, st>>>(z, B.av, B.pv,
        LANDC, DH, LANDC, 1.f, S, (long long)LANDC * DH, (long long)LANDC * DH);
  cvt_bf16<<<(HEADS * LANDC * DH) / 256, 256, 0, st>>>(B.pv, B.pvb, (long long)HEADS * LANDC * DH);
  gemmL(st, B.a1b, B.pvb, B.attn, nullptr, nullptr, NPAD, DH, LANDC,
        (long long)NPAD * LANDC, (long long)LANDC * DH, (long long)NPAD * DH, HEADS, 1, 0);
  // + depthwise residual conv, merge heads, project, residual add
  res_merge<<<(NPAD * DIMC) / 256, 256, 0, st>>>(B.attn, B.vh, resw, B.attnb);
  gemmL(st, B.attnb, wout, B.proj, nullptr, outb, NPAD, DIMC, DIMC, 0, 0, 0, 1, 0, 0);
  resid_tail<<<(NTOK * DIMC) / 256, 256, 0, st>>>(tok, B.proj);
}

extern "C" void kernel_launch(void* const* d_in, const int* in_sizes, int n_in,
                              void* d_out, int out_size, void* d_ws, size_t ws_size,
                              hipStream_t stream) {
  (void)in_sizes; (void)n_in; (void)out_size; (void)ws_size;
  const float* x      = (const float*)d_in[0];
  const float* fc1w   = (const float*)d_in[1];
  const float* fc1b   = (const float*)d_in[2];
  const float* cls    = (const float*)d_in[3];
  const float* l1ng   = (const float*)d_in[4];
  const float* l1nb   = (const float*)d_in[5];
  const float* l1qkvw = (const float*)d_in[6];
  const float* l1outw = (const float*)d_in[7];
  const float* l1outb = (const float*)d_in[8];
  const float* l1resw = (const float*)d_in[9];
  const float* l2ng   = (const float*)d_in[10];
  const float* l2nb   = (const float*)d_in[11];
  const float* l2qkvw = (const float*)d_in[12];
  const float* l2outw = (const float*)d_in[13];
  const float* l2outb = (const float*)d_in[14];
  const float* l2resw = (const float*)d_in[15];
  const float* p7w    = (const float*)d_in[16];
  const float* p7b    = (const float*)d_in[17];
  const float* p5w    = (const float*)d_in[18];
  const float* p5b    = (const float*)d_in[19];
  const float* p3w    = (const float*)d_in[20];
  const float* p3b    = (const float*)d_in[21];
  const float* normg  = (const float*)d_in[22];
  const float* normb  = (const float*)d_in[23];
  const float* fc2w   = (const float*)d_in[24];
  const float* fc2b   = (const float*)d_in[25];
  float* out = (float*)d_out;

  char* base = (char*)d_ws;
  size_t off = 0;
  auto alloc = [&](size_t bytes) -> void* {
    void* p = base + off;
    off += (bytes + 255) & ~(size_t)255;
    return p;
  };
  __bf16* xbf   = (__bf16*)alloc((size_t)NSEQ * KIN * 2);
  __bf16* wfc1  = (__bf16*)alloc((size_t)DIMC * KIN * 2);
  __bf16* wqkv1 = (__bf16*)alloc((size_t)3 * DIMC * DIMC * 2);
  __bf16* wout1 = (__bf16*)alloc((size_t)DIMC * DIMC * 2);
  __bf16* wqkv2 = (__bf16*)alloc((size_t)3 * DIMC * DIMC * 2);
  __bf16* wout2 = (__bf16*)alloc((size_t)DIMC * DIMC * 2);
  float*  hbuf  = (float*)alloc((size_t)NSEQ * DIMC * 4);
  float*  tokA  = (float*)alloc((size_t)NTOK * DIMC * 4);
  float*  tokB  = (float*)alloc((size_t)NTOK * DIMC * 4);

  LayerBufs LB;
  LB.xln  = (__bf16*)alloc((size_t)NPAD * DIMC * 2);
  LB.Sbig = (float*)alloc((size_t)HEADS * NPAD * LANDC * 4);
  LB.qkvo = (__bf16*)LB.Sbig;                 // alias: qkv output dead before logits
  LB.qh   = (__bf16*)alloc((size_t)HEADS * NPAD * DH * 2);
  LB.kh   = (__bf16*)alloc((size_t)HEADS * NPAD * DH * 2);
  LB.vh   = (__bf16*)alloc((size_t)HEADS * NPAD * DH * 2);
  LB.qlb  = (__bf16*)alloc((size_t)HEADS * LANDC * DH * 2);
  LB.klb  = (__bf16*)alloc((size_t)HEADS * LANDC * DH * 2);
  LB.a1b  = (__bf16*)alloc((size_t)HEADS * NPAD * LANDC * 2);
  LB.a3b  = (__bf16*)alloc((size_t)HEADS * NPAD * LANDC * 2);
  LB.s2   = (float*)alloc((size_t)HEADS * LANDC * LANDC * 4);
  LB.av   = (float*)alloc((size_t)HEADS * LANDC * DH * 4);
  LB.pv   = (float*)alloc((size_t)HEADS * LANDC * DH * 4);
  LB.pvb  = (__bf16*)alloc((size_t)HEADS * LANDC * DH * 2);
  LB.attn = (float*)alloc((size_t)HEADS * NPAD * DH * 4);
  LB.proj = LB.attn;                          // alias: attn dead before projection
  LB.attnb = (__bf16*)hbuf;                   // alias: fc1 staging dead after assemble
  LB.gmax = (float*)alloc(256);
  // pinv temporaries alias the (dead after fc1) bf16 input buffer
  LB.z0 = (float*)xbf;
  LB.z1 = LB.z0 + (size_t)HEADS * LANDC * LANDC;
  LB.xz = LB.z1 + (size_t)HEADS * LANDC * LANDC;
  LB.tb = LB.xz + (size_t)HEADS * LANDC * LANDC;
  LB.ub = LB.tb + (size_t)HEADS * LANDC * LANDC;

  // ---- prologue: convert inputs/weights, fc1+relu, token assembly ----
  cvt_bf16<<<((long long)NSEQ * KIN) / 256, 256, 0, stream>>>(x, xbf, (long long)NSEQ * KIN);
  cvt_bf16<<<(DIMC * KIN) / 256, 256, 0, stream>>>(fc1w, wfc1, (long long)DIMC * KIN);
  cvt_bf16<<<(3 * DIMC * DIMC) / 256, 256, 0, stream>>>(l1qkvw, wqkv1, (long long)3 * DIMC * DIMC);
  cvt_bf16<<<(DIMC * DIMC) / 256, 256, 0, stream>>>(l1outw, wout1, (long long)DIMC * DIMC);
  cvt_bf16<<<(3 * DIMC * DIMC) / 256, 256, 0, stream>>>(l2qkvw, wqkv2, (long long)3 * DIMC * DIMC);
  cvt_bf16<<<(DIMC * DIMC) / 256, 256, 0, stream>>>(l2outw, wout2, (long long)DIMC * DIMC);
  gemmL(stream, xbf, wfc1, hbuf, nullptr, fc1b, NSEQ, DIMC, KIN, 0, 0, 0, 1, 0, 1);
  assemble_tok<<<(NTOK * DIMC) / 256, 256, 0, stream>>>(hbuf, cls, tokA);
  // ---- two transformer layers with PPEG in between ----
  run_layer(stream, tokA, l1ng, l1nb, wqkv1, wout1, l1outb, l1resw, LB);
  ppeg_k<<<(NTOK * DIMC) / 256, 256, 0, stream>>>(tokA, tokB, p7w, p7b, p5w, p5b, p3w, p3b);
  run_layer(stream, tokB, l2ng, l2nb, wqkv2, wout2, l2outb, l2resw, LB);
  // ---- final layernorm(cls) + fc2 ----
  final_head<<<1, 256, 0, stream>>>(tokB, normg, normb, fc2w, fc2b, out);
}
